// MyMultiHeadAttention_10093173145970
// MI455X (gfx1250) — matrix operand order repaired and checked by reading, not compiled, not run on hardware
//
#include <hip/hip_runtime.h>
#include <hip/hip_bf16.h>

typedef __attribute__((ext_vector_type(16))) __bf16 v16bf;
typedef __attribute__((ext_vector_type(8)))  __bf16 v8bf;
typedef __attribute__((ext_vector_type(8)))  float  v8f;

#define B_  2
#define S_  2048
#define D_  1024
#define H_  16
#define DK_ 64

__device__ __forceinline__ v8f wmma_bf16(v16bf a, v16bf b, v8f c) {
  // (neg_a, A, neg_b, B, c_mod, C, reuse_a, reuse_b)
  return __builtin_amdgcn_wmma_f32_16x16x32_bf16(false, a, false, b, (short)0, c, false, false);
}

__device__ __forceinline__ v8f zero8() {
  v8f z;
#pragma unroll
  for (int i = 0; i < 8; ++i) z[i] = 0.f;
  return z;
}

// A fragment 16x32 bf16 from row-major [16 x >=kbase+32], row stride ld.
// Half 0: K {kbase+0..7, kbase+16..23}; half 1: K {kbase+8..15, kbase+24..31}.
__device__ __forceinline__ v16bf frag_A(const __bf16* base, int ld, int kbase, int lane) {
  int m = lane & 15, h = (lane >> 4) & 1;
  const __bf16* p = base + (size_t)m * ld + kbase + h * 8;
  v8bf lo = *(const v8bf*)p;
  v8bf hi = *(const v8bf*)(p + 16);
  v16bf r;
#pragma unroll
  for (int i = 0; i < 8; ++i) { r[i] = lo[i]; r[i + 8] = hi[i]; }
  return r;
}

// B fragment 32x16 bf16 from K-transposed storage Bt[n][k] (row = output col n),
// row stride ld. Half 0: K {kbase+0..7, kbase+16..23}; half 1: K {kbase+8..15, kbase+24..31}.
__device__ __forceinline__ v16bf frag_B(const __bf16* base, int ld, int kbase, int lane) {
  int n = lane & 15, h = (lane >> 4) & 1;
  const __bf16* p = base + (size_t)n * ld + kbase + h * 8;
  v8bf lo = *(const v8bf*)p;
  v8bf hi = *(const v8bf*)(p + 16);
  v16bf r;
#pragma unroll
  for (int i = 0; i < 8; ++i) { r[i] = lo[i]; r[i + 8] = hi[i]; }
  return r;
}

__device__ __forceinline__ float red_max16(float v) {
#pragma unroll
  for (int o = 1; o < 16; o <<= 1) v = fmaxf(v, __shfl_xor(v, o, 32));
  return v;
}
__device__ __forceinline__ float red_sum16(float v) {
#pragma unroll
  for (int o = 1; o < 16; o <<= 1) v += __shfl_xor(v, o, 32);
  return v;
}

// ---------- conversion / layout kernels ----------
__global__ void cvt_bf16_kernel(const float* __restrict__ in, __bf16* __restrict__ out, int n) {
  int i = blockIdx.x * blockDim.x + threadIdx.x;
  if (i < n) out[i] = (__bf16)in[i];
}

// [G,R,C] f32 -> [G,C,R] bf16
__global__ void transpose_bf16_kernel(const float* __restrict__ in, __bf16* __restrict__ out,
                                      int G, int R, int C) {
  int i = blockIdx.x * blockDim.x + threadIdx.x;
  int n = G * R * C;
  if (i >= n) return;
  int g = i / (R * C);
  int rc = i - g * (R * C);
  int r = rc / C, c = rc - r * C;
  out[((size_t)g * C + c) * R + r] = (__bf16)in[i];
}

// ---------- QKV projection + RoPE ----------
// One wave -> one 16(s) x 64(dk) tile of one of {q,k,v} for one (b,h).
__global__ void __launch_bounds__(256) qkv_rope_kernel(
    const __bf16* __restrict__ xb,  const __bf16* __restrict__ Wqt,
    const __bf16* __restrict__ Wkt, const __bf16* __restrict__ Wvt,
    __bf16* __restrict__ Qb, __bf16* __restrict__ Kb, __bf16* __restrict__ Vt) {
  int lane = threadIdx.x & 31;
  int wid  = blockIdx.x * (blockDim.x >> 5) + (threadIdx.x >> 5);
  int which = wid % 3;             // 0=q 1=k 2=v
  int rest  = wid / 3;
  int mt    = rest % (S_ / 16);
  int bh    = rest / (S_ / 16);
  int b = bh / H_, h = bh % H_;
  int s_base = mt * 16;

  const __bf16* W  = (which == 0) ? Wqt : ((which == 1) ? Wkt : Wvt);
  const __bf16* A  = xb + ((size_t)b * S_ + s_base) * D_;
  const __bf16* Bw = W + (size_t)h * DK_ * D_;

  v8f acc[4];
#pragma unroll
  for (int t = 0; t < 4; ++t) acc[t] = zero8();

  for (int kb = 0; kb < D_; kb += 32) {
    v16bf a = frag_A(A, D_, kb, lane);
#pragma unroll
    for (int t = 0; t < 4; ++t) {
      v16bf bf = frag_B(Bw + (size_t)(t * 16) * D_, D_, kb, lane);
      acc[t] = wmma_bf16(a, bf, acc[t]);
    }
  }

  int l15 = lane & 15, half = (lane >> 4) & 1;
  if (which < 2) {  // RoPE on q,k: pairs (2i,2i+1) live in adjacent lanes -> shfl_xor 1
    const float LOG_THETA = 9.210340371976184f;  // ln(10000)
#pragma unroll
    for (int t = 0; t < 4; ++t) {
#pragma unroll
      for (int r = 0; r < 8; ++r) {
        int srow = s_base + r + 8 * half;
        int col  = t * 16 + l15;
        float invf = __expf(-((float)(col & ~1) / (float)DK_) * LOG_THETA);
        float ang  = (float)srow * invf;
        float cv = __cosf(ang), sv = __sinf(ang);
        float v  = acc[t][r];
        float pv = __shfl_xor(v, 1, 32);  // partner within the rotation pair
        acc[t][r] = (col & 1) ? (v * cv + pv * sv) : (v * cv - pv * sv);
      }
    }
  }

  __bf16* Qp = Qb + (size_t)bh * S_ * DK_;
  __bf16* Kp = Kb + (size_t)bh * S_ * DK_;
  __bf16* Vp = Vt + (size_t)bh * DK_ * S_;  // V stored transposed [DK, S]
#pragma unroll
  for (int t = 0; t < 4; ++t) {
#pragma unroll
    for (int r = 0; r < 8; ++r) {
      int srow = s_base + r + 8 * half;
      int col  = t * 16 + l15;
      __bf16 val = (__bf16)acc[t][r];
      if (which == 0)      Qp[(size_t)srow * DK_ + col] = val;
      else if (which == 1) Kp[(size_t)srow * DK_ + col] = val;
      else                 Vp[(size_t)col * S_ + srow] = val;
    }
  }
}

// ---------- flash attention (causal, online softmax) ----------
// One wave -> 16 query rows x DK=64 for one (b,h). Keys processed 32 at a time.
__global__ void __launch_bounds__(256) flash_attn_kernel(
    const __bf16* __restrict__ Qb, const __bf16* __restrict__ Kb,
    const __bf16* __restrict__ Vt, __bf16* __restrict__ AO) {
  __shared__ __bf16 sP[8][16 * 32];   // per-wave 16x32 P tile (C-layout -> A-layout bounce)
  int lane = threadIdx.x & 31;
  int wi   = threadIdx.x >> 5;
  int wid  = blockIdx.x * 8 + wi;
  int qt   = wid % (S_ / 16);
  int bh   = wid / (S_ / 16);
  int b = bh / H_, h = bh % H_;
  int q_base = qt * 16;
  int l15 = lane & 15, half = (lane >> 4) & 1;

  const __bf16* Qp = Qb + (size_t)bh * S_ * DK_ + (size_t)q_base * DK_;
  const __bf16* Kp = Kb + (size_t)bh * S_ * DK_;
  const __bf16* Vp = Vt + (size_t)bh * DK_ * S_;

  v16bf aq0 = frag_A(Qp, DK_, 0, lane);
  v16bf aq1 = frag_A(Qp, DK_, 32, lane);

  v8f acc[4];
#pragma unroll
  for (int t = 0; t < 4; ++t) acc[t] = zero8();
  float m_r[8], l_r[8];
#pragma unroll
  for (int r = 0; r < 8; ++r) { m_r[r] = -1e30f; l_r[r] = 0.f; }

  __bf16* lds = &sP[wi][0];
  int nkb = (q_base + 16 + 31) >> 5;  // causal: keys 0 .. q_base+15

  for (int kbi = 0; kbi < nkb; ++kbi) {
    int kb = kbi * 32;
    // scores: Q(16x64) x K^T -> two 16x16 tiles (keys kb..+15, kb+16..+31)
    v16bf b00 = frag_B(Kp + (size_t)kb * DK_,        DK_, 0,  lane);
    v16bf b01 = frag_B(Kp + (size_t)kb * DK_,        DK_, 32, lane);
    v16bf b10 = frag_B(Kp + (size_t)(kb + 16) * DK_, DK_, 0,  lane);
    v16bf b11 = frag_B(Kp + (size_t)(kb + 16) * DK_, DK_, 32, lane);
    v8f s0 = zero8(), s1 = zero8();
    s0 = wmma_bf16(aq0, b00, s0); s0 = wmma_bf16(aq1, b01, s0);
    s1 = wmma_bf16(aq0, b10, s1); s1 = wmma_bf16(aq1, b11, s1);

#pragma unroll
    for (int r = 0; r < 8; ++r) {
      int qrow = q_base + r + 8 * half;
      float v0 = s0[r] * 0.125f; if (kb + l15 > qrow)      v0 = -1e30f;
      float v1 = s1[r] * 0.125f; if (kb + 16 + l15 > qrow) v1 = -1e30f;
      float mx = red_max16(fmaxf(v0, v1));
      float mn = fmaxf(m_r[r], mx);
      float corr = __expf(m_r[r] - mn);
      float p0 = __expf(v0 - mn);
      float p1 = __expf(v1 - mn);
      float rs = red_sum16(p0 + p1);
      l_r[r] = l_r[r] * corr + rs;
      m_r[r] = mn;
#pragma unroll
      for (int t = 0; t < 4; ++t) acc[t][r] *= corr;
      lds[(r + 8 * half) * 32 + l15]      = (__bf16)p0;
      lds[(r + 8 * half) * 32 + 16 + l15] = (__bf16)p1;
    }
    asm volatile("s_wait_dscnt 0" ::: "memory");  // P tile visible wave-wide
    v16bf ap = frag_A(lds, 32, 0, lane);          // 16x32 P as A fragment
#pragma unroll
    for (int t = 0; t < 4; ++t) {
      v16bf bv = frag_B(Vp + (size_t)(t * 16) * S_, S_, kb, lane);  // V^T contiguous
      acc[t] = wmma_bf16(ap, bv, acc[t]);
    }
  }

  // normalize + store concat-head layout [B, S, H*DK] (bf16)
#pragma unroll
  for (int t = 0; t < 4; ++t) {
#pragma unroll
    for (int r = 0; r < 8; ++r) {
      int srow = q_base + r + 8 * half;
      float o = acc[t][r] / l_r[r];
      AO[((size_t)b * S_ + srow) * D_ + h * DK_ + t * 16 + l15] = (__bf16)o;
    }
  }
}

// ---------- output projection: [B*S,1024] x Wo[1024,1024] -> f32 ----------
__global__ void __launch_bounds__(256) out_proj_kernel(
    const __bf16* __restrict__ AO, const __bf16* __restrict__ Wot, float* __restrict__ out) {
  int lane = threadIdx.x & 31;
  int wid  = blockIdx.x * 8 + (threadIdx.x >> 5);
  int nt = wid % (D_ / 64);   // 16 output-column chunks of 64
  int mt = wid / (D_ / 64);   // 256 row tiles of 16
  const __bf16* A = AO + (size_t)mt * 16 * D_;

  v8f acc[4];
#pragma unroll
  for (int t = 0; t < 4; ++t) acc[t] = zero8();

  for (int kb = 0; kb < D_; kb += 32) {
    v16bf a = frag_A(A, D_, kb, lane);
#pragma unroll
    for (int t = 0; t < 4; ++t) {
      v16bf bw = frag_B(Wot + (size_t)(nt * 64 + t * 16) * D_, D_, kb, lane);
      acc[t] = wmma_bf16(a, bw, acc[t]);
    }
  }
  int l15 = lane & 15, half = (lane >> 4) & 1;
#pragma unroll
  for (int t = 0; t < 4; ++t) {
#pragma unroll
    for (int r = 0; r < 8; ++r) {
      size_t row = (size_t)mt * 16 + r + 8 * half;
      out[row * D_ + nt * 64 + t * 16 + l15] = acc[t][r];
    }
  }
}

extern "C" void kernel_launch(void* const* d_in, const int* in_sizes, int n_in,
                              void* d_out, int out_size, void* d_ws, size_t ws_size,
                              hipStream_t stream) {
  (void)in_sizes; (void)n_in; (void)out_size; (void)ws_size;
  const float* x  = (const float*)d_in[0];
  const float* Wq = (const float*)d_in[1];
  const float* Wk = (const float*)d_in[2];
  const float* Wv = (const float*)d_in[3];
  const float* Wo = (const float*)d_in[4];

  __bf16* ws = (__bf16*)d_ws;
  size_t off = 0;
  __bf16* xb  = ws + off; off += (size_t)B_ * S_ * D_;      // 4M
  __bf16* Wqt = ws + off; off += (size_t)H_ * DK_ * D_;     // 1M
  __bf16* Wkt = ws + off; off += (size_t)H_ * DK_ * D_;
  __bf16* Wvt = ws + off; off += (size_t)H_ * DK_ * D_;
  __bf16* Wot = ws + off; off += (size_t)D_ * D_;
  __bf16* Qb  = ws + off; off += (size_t)B_ * H_ * S_ * DK_;
  __bf16* Kb  = ws + off; off += (size_t)B_ * H_ * S_ * DK_;
  __bf16* Vt  = ws + off; off += (size_t)B_ * H_ * DK_ * S_;
  __bf16* AO  = ws + off; off += (size_t)B_ * S_ * D_;      // total 24M bf16 = 48MB

  int nx = B_ * S_ * D_;
  cvt_bf16_kernel<<<(nx + 255) / 256, 256, 0, stream>>>(x, xb, nx);
  int nw = H_ * D_ * DK_;
  transpose_bf16_kernel<<<(nw + 255) / 256, 256, 0, stream>>>(Wq, Wqt, H_, D_, DK_);
  transpose_bf16_kernel<<<(nw + 255) / 256, 256, 0, stream>>>(Wk, Wkt, H_, D_, DK_);
  transpose_bf16_kernel<<<(nw + 255) / 256, 256, 0, stream>>>(Wv, Wvt, H_, D_, DK_);
  int no = D_ * D_;
  transpose_bf16_kernel<<<(no + 255) / 256, 256, 0, stream>>>(Wo, Wot, 1, D_, D_);

  qkv_rope_kernel<<<(3 * B_ * H_ * (S_ / 16)) / 8, 256, 0, stream>>>(
      xb, Wqt, Wkt, Wvt, Qb, Kb, Vt);
  flash_attn_kernel<<<(B_ * H_ * (S_ / 16)) / 8, 256, 0, stream>>>(Qb, Kb, Vt, AO);
  out_proj_kernel<<<((B_ * S_ / 16) * (D_ / 64)) / 8, 256, 0, stream>>>(
      AO, Wot, (float*)d_out);
}
